// SparseAttention_19628000543054
// MI455X (gfx1250) — matrix operand order repaired and checked by reading, not compiled, not run on hardware
//
#include <hip/hip_runtime.h>
#include <hip/hip_bf16.h>

// Sparse attention, MI455X (gfx1250, wave32, WMMA).
// Pipeline: cast/transpose -> QKV GEMM (bf16 WMMA) -> QK^T scores (WMMA)
//        -> exact kth-smallest threshold (4-pass radix select)
//        -> masked softmax (writes P bf16 in place over scores)
//        -> P@V (WMMA) -> output GEMM (WMMA) + bias -> f32 d_out.
// Each GEMM wave computes a 16x64 tile; operand fragments are explicitly
// double-buffered so next K-step loads overlap the current 4-WMMA chain.

#define BATCH 4
#define SEQ   1024
#define DMODEL 768
#define NH    12
#define HD    64
#define BH    (BATCH*NH)      /* 48 */
#define KTH   104857          /* int(1024*1024*0.1) */
#define SPLIT 32

typedef __bf16 bf16;
typedef __attribute__((ext_vector_type(16))) __bf16 v16bf;
typedef __attribute__((ext_vector_type(8)))  __bf16 v8bf;
typedef __attribute__((ext_vector_type(8)))  float  v8f;

__device__ __forceinline__ bf16 to_bf16(float f) {
  unsigned u = __float_as_uint(f);
  u += 0x7FFFu + ((u >> 16) & 1u);            // round-to-nearest-even
  unsigned short h = (unsigned short)(u >> 16);
  bf16 r; __builtin_memcpy(&r, &h, 2); return r;
}

__device__ __forceinline__ unsigned orderKey(float f) {
  unsigned u = __float_as_uint(f);
  return (u & 0x80000000u) ? ~u : (u ^ 0x80000000u);
}
__device__ __forceinline__ float keyToFloat(unsigned k) {
  unsigned u = (k & 0x80000000u) ? (k ^ 0x80000000u) : ~k;
  return __uint_as_float(u);
}

__device__ __forceinline__ v16bf load_afrag(const bf16* __restrict__ arow,
                                            int k0, int half) {
  const bf16* pa = arow + k0 + half * 8;
  v8bf lo = *(const v8bf*)pa;                  // K = k0+half*8    .. +7
  v8bf hi = *(const v8bf*)(pa + 16);           // K = k0+16+half*8 .. +7
  return __builtin_shufflevector(lo, hi,
      0,1,2,3,4,5,6,7,8,9,10,11,12,13,14,15);
}
__device__ __forceinline__ v16bf load_bfrag(const bf16* __restrict__ brow,
                                            int k0, int half) {
  const bf16* pb = brow + k0 + half * 8;
  v8bf lo = *(const v8bf*)pb;                  // K = k0+half*8    .. +7
  v8bf hi = *(const v8bf*)(pb + 16);           // K = k0+16+half*8 .. +7
  return __builtin_shufflevector(lo, hi,
      0,1,2,3,4,5,6,7,8,9,10,11,12,13,14,15);
}

// 16x64 output tile per wave: 4 accumulators; A/B fragments double-buffered.
// A: row-major [M,K] bf16 (lda); BT: row-major [Ncols,K] bf16 (ldb).
__device__ __forceinline__ void wmma_tile4(const bf16* __restrict__ A, int lda,
                                           const bf16* __restrict__ BT, int ldb,
                                           int row0, int col0, int K, v8f acc[4]) {
  const int lane = threadIdx.x & 31;
  const int half = lane >> 4;          // K-half select
  const int m    = lane & 15;          // A row / B column within tile
  const bf16* arow = A + (size_t)(row0 + m) * lda;
  const bf16* brow[4];
#pragma unroll
  for (int j = 0; j < 4; ++j)
    brow[j] = BT + (size_t)(col0 + 16 * j + m) * ldb;

  v16bf a0 = load_afrag(arow, 0, half);
  v16bf b0[4];
#pragma unroll
  for (int j = 0; j < 4; ++j) b0[j] = load_bfrag(brow[j], 0, half);

  for (int k0 = 0; k0 < K; k0 += 32) {
    // next K-step (dummy in-bounds reload of k=0 on the last iteration keeps
    // control flow uniform; values are discarded)
    int kn = (k0 + 32 < K) ? (k0 + 32) : 0;
    __builtin_prefetch(arow + k0 + 128, 0, 3);
    v16bf a1 = load_afrag(arow, kn, half);
    v16bf b1[4];
#pragma unroll
    for (int j = 0; j < 4; ++j) b1[j] = load_bfrag(brow[j], kn, half);
#pragma unroll
    for (int j = 0; j < 4; ++j)
      acc[j] = __builtin_amdgcn_wmma_f32_16x16x32_bf16(
          false, a0, false, b0[j], (short)0, acc[j], false, false);
    a0 = a1;
#pragma unroll
    for (int j = 0; j < 4; ++j) b0[j] = b1[j];
  }
}

// ---------------- casts ----------------
__global__ void k_cast(const float* __restrict__ in, bf16* __restrict__ out, int n) {
  int i = blockIdx.x * blockDim.x + threadIdx.x;
  if (i < n) out[i] = to_bf16(in[i]);
}
// in: [rows,cols] f32 row-major -> out: [cols,rows] bf16 row-major
__global__ void k_cast_T(const float* __restrict__ in, bf16* __restrict__ out,
                         int rows, int cols) {
  int i = blockIdx.x * blockDim.x + threadIdx.x;
  if (i < rows * cols) {
    int r = i / cols, c = i - r * cols;
    out[(size_t)c * rows + r] = to_bf16(in[i]);
  }
}

// ---------------- QKV GEMM ----------------
// [4096,2304] = xb[4096,768] @ WqkvT'[2304,768]; scatter to q,k (per-head) and vT.
__global__ void k_gemm_qkv(const bf16* __restrict__ A, const bf16* __restrict__ BT,
                           const float* __restrict__ bias,
                           bf16* __restrict__ q, bf16* __restrict__ kk,
                           bf16* __restrict__ vT) {
  const int NG = (3 * DMODEL) / 64;                 // 36 column groups of 64
  int tile = blockIdx.x * 4 + (threadIdx.x >> 5);   // 256*36 = 9216 waves
  int tm = tile / NG, tg = tile - tm * NG;
  int row0 = tm * 16, col0 = tg * 64;
  v8f acc[4] = {};
  wmma_tile4(A, DMODEL, BT, DMODEL, row0, col0, DMODEL, acc);
  int lane = threadIdx.x & 31;
  int half = lane >> 4, n = lane & 15;
#pragma unroll
  for (int j = 0; j < 4; ++j) {
    int c = col0 + 16 * j + n;
    int which = c / DMODEL;
    int rem = c - which * DMODEL;
    int h = rem >> 6, d = rem & 63;
    float bv = bias[c];
#pragma unroll
    for (int r = 0; r < 8; ++r) {
      int mrow = row0 + r + half * 8;
      int b = mrow >> 10, nn = mrow & 1023;
      bf16 val = to_bf16(acc[j][r] + bv);
      size_t bhh = (size_t)(b * NH + h);
      if (which == 0)      q [(bhh * SEQ + nn) * HD + d] = val;
      else if (which == 1) kk[(bhh * SEQ + nn) * HD + d] = val;
      else                 vT[(bhh * HD + d) * SEQ + nn] = val;  // transposed V
    }
  }
}

// ---------------- scores = Q K^T * 1/sqrt(hd) ----------------
__global__ void k_gemm_scores(const bf16* __restrict__ q, const bf16* __restrict__ kk,
                              float* __restrict__ scores) {
  int tile = blockIdx.x * 4 + (threadIdx.x >> 5);   // 48 * 64 * 16 waves
  int bh = tile >> 10;
  int t  = tile & 1023;
  int tm = t >> 4, tg = t & 15;
  const bf16* A  = q  + (size_t)bh * SEQ * HD;
  const bf16* BT = kk + (size_t)bh * SEQ * HD;      // K matrix is already [N,hd]=B^T
  v8f acc[4] = {};
  wmma_tile4(A, HD, BT, HD, tm * 16, tg * 64, HD, acc);
  int lane = threadIdx.x & 31;
  int half = lane >> 4, n = lane & 15;
  float* out = scores + (size_t)bh * SEQ * SEQ;
#pragma unroll
  for (int j = 0; j < 4; ++j)
#pragma unroll
    for (int r = 0; r < 8; ++r)
      out[(size_t)(tm * 16 + r + half * 8) * SEQ + tg * 64 + 16 * j + n] =
          acc[j][r] * 0.125f;
}

// ---------------- exact kth-smallest per (b,h): radix select ----------------
__global__ void k_sel_init(unsigned* hist, unsigned* prefix, unsigned* krem) {
  int i = blockIdx.x * blockDim.x + threadIdx.x;
  if (i < BH * 256) hist[i] = 0;
  if (i < BH) { prefix[i] = 0u; krem[i] = KTH; }
}

__global__ void k_sel_hist(const float* __restrict__ scores,
                           const unsigned* __restrict__ prefix,
                           unsigned* __restrict__ hist, int pass) {
  int bh   = blockIdx.x / SPLIT;
  int part = blockIdx.x - bh * SPLIT;
  __shared__ unsigned lh[256];
  int tid = threadIdx.x;
  lh[tid] = 0;
  __syncthreads();
  int shift = 24 - 8 * pass;
  unsigned pmask = pass ? (0xFFFFFFFFu << (shift + 8)) : 0u;
  unsigned pref  = prefix[bh];
  const float* s = scores + (size_t)bh * SEQ * SEQ;
  const int chunk = SEQ * SEQ / SPLIT;
  int base = part * chunk;
  for (int i = tid; i < chunk; i += 256) {
    unsigned key = orderKey(s[base + i]);
    if ((key & pmask) == pref) atomicAdd(&lh[(key >> shift) & 255u], 1u);
  }
  __syncthreads();
  if (lh[tid]) atomicAdd(&hist[bh * 256 + tid], lh[tid]);
}

__global__ void k_sel_pick(unsigned* __restrict__ hist, unsigned* __restrict__ prefix,
                           unsigned* __restrict__ krem, float* __restrict__ thr,
                           int pass) {
  int bh = blockIdx.x, tid = threadIdx.x;
  __shared__ unsigned h[256];
  h[tid] = hist[bh * 256 + tid];
  __syncthreads();
  if (tid == 0) {
    unsigned k = krem[bh], cum = 0; int bin = 0;
    for (; bin < 256; ++bin) {
      if (cum + h[bin] >= k) break;
      cum += h[bin];
    }
    int shift = 24 - 8 * pass;
    unsigned np = prefix[bh] | ((unsigned)bin << shift);
    prefix[bh] = np;
    krem[bh]   = k - cum;
    if (pass == 3) thr[bh] = keyToFloat(np);   // exact sorted[kth-1]
  }
  hist[bh * 256 + tid] = 0;                    // ready for next pass
}

// ---------------- masked softmax; P written bf16 in place ----------------
__global__ void k_softmax(float* __restrict__ scores, const float* __restrict__ thr) {
  int row = blockIdx.x;                 // 0 .. BH*SEQ-1
  int bh  = row >> 10;
  float t = thr[bh];
  float* s = scores + (size_t)row * SEQ;
  int tid = threadIdx.x;                // 256 threads
  float v[4];
#pragma unroll
  for (int i = 0; i < 4; ++i) {
    float x = s[i * 256 + tid];
    v[i] = (x <= t) ? -1e9f : x;        // reference mask semantics
  }
  __shared__ float red[256];
  float m = fmaxf(fmaxf(v[0], v[1]), fmaxf(v[2], v[3]));
  red[tid] = m; __syncthreads();
  for (int st = 128; st > 0; st >>= 1) {
    if (tid < st) red[tid] = fmaxf(red[tid], red[tid + st]);
    __syncthreads();
  }
  m = red[0]; __syncthreads();
  float e[4], sum = 0.f;
#pragma unroll
  for (int i = 0; i < 4; ++i) { e[i] = __expf(v[i] - m); sum += e[i]; }
  red[tid] = sum; __syncthreads();
  for (int st = 128; st > 0; st >>= 1) {
    if (tid < st) red[tid] += red[tid + st];
    __syncthreads();
  }
  float inv = 1.0f / red[0];
  bf16* P = (bf16*)scores + (size_t)row * (2 * SEQ);   // row stride 2048 bf16
#pragma unroll
  for (int i = 0; i < 4; ++i) P[i * 256 + tid] = to_bf16(e[i] * inv);
}

// ---------------- O = P @ V (per head), scatter to [B,N,D] bf16 ----------------
// One wave computes a full 16-row x 64-col (= whole head width) tile.
__global__ void k_gemm_attnout(const float* __restrict__ scoresP,
                               const bf16* __restrict__ vT,
                               bf16* __restrict__ obuf) {
  int tile = blockIdx.x * 4 + (threadIdx.x >> 5);  // 48 * 64 waves
  int bh = tile >> 6;
  int tm = tile & 63;
  const bf16* A  = (const bf16*)scoresP + (size_t)bh * SEQ * (2 * SEQ);
  const bf16* BT = vT + (size_t)bh * HD * SEQ;
  v8f acc[4] = {};
  wmma_tile4(A, 2 * SEQ, BT, SEQ, tm * 16, 0, SEQ, acc);
  int b = bh / NH, h = bh - b * NH;
  int lane = threadIdx.x & 31;
  int half = lane >> 4, n = lane & 15;
#pragma unroll
  for (int j = 0; j < 4; ++j) {
    int d = 16 * j + n;
#pragma unroll
    for (int r = 0; r < 8; ++r) {
      int nn = tm * 16 + r + half * 8;
      obuf[((size_t)(b * SEQ + nn)) * DMODEL + h * HD + d] = to_bf16(acc[j][r]);
    }
  }
}

// ---------------- Y = O @ W_out + b_out (f32 out) ----------------
__global__ void k_gemm_final(const bf16* __restrict__ A, const bf16* __restrict__ BT,
                             const float* __restrict__ bias, float* __restrict__ out) {
  const int NG = DMODEL / 64;                      // 12 column groups
  int tile = blockIdx.x * 4 + (threadIdx.x >> 5);  // 256*12 = 3072 waves
  int tm = tile / NG, tg = tile - tm * NG;
  v8f acc[4] = {};
  wmma_tile4(A, DMODEL, BT, DMODEL, tm * 16, tg * 64, DMODEL, acc);
  int lane = threadIdx.x & 31;
  int half = lane >> 4, n = lane & 15;
#pragma unroll
  for (int j = 0; j < 4; ++j) {
    int c = tg * 64 + 16 * j + n;
    float bv = bias[c];
#pragma unroll
    for (int r = 0; r < 8; ++r)
      out[(size_t)(tm * 16 + r + half * 8) * DMODEL + c] = acc[j][r] + bv;
  }
}

extern "C" void kernel_launch(void* const* d_in, const int* in_sizes, int n_in,
                              void* d_out, int out_size, void* d_ws, size_t ws_size,
                              hipStream_t stream) {
  (void)in_sizes; (void)n_in; (void)out_size; (void)ws_size;
  const float* x    = (const float*)d_in[0];
  const float* Wqkv = (const float*)d_in[1];
  const float* bqkv = (const float*)d_in[2];
  const float* Wout = (const float*)d_in[3];
  const float* bout = (const float*)d_in[4];
  float* out = (float*)d_out;

  char* ws = (char*)d_ws;
  auto carve = [&](size_t bytes) {
    char* p = ws; ws += (bytes + 255) & ~(size_t)255; return p;
  };
  bf16* xb     = (bf16*)carve((size_t)BATCH * SEQ * DMODEL * 2);
  bf16* wqkvT  = (bf16*)carve((size_t)3 * DMODEL * DMODEL * 2);
  bf16* woutT  = (bf16*)carve((size_t)DMODEL * DMODEL * 2);
  bf16* qb     = (bf16*)carve((size_t)BH * SEQ * HD * 2);
  bf16* kb     = (bf16*)carve((size_t)BH * SEQ * HD * 2);
  bf16* vtb    = (bf16*)carve((size_t)BH * HD * SEQ * 2);
  bf16* ob     = (bf16*)carve((size_t)BATCH * SEQ * DMODEL * 2);
  unsigned* hist   = (unsigned*)carve((size_t)BH * 256 * 4);
  unsigned* prefix = (unsigned*)carve((size_t)BH * 4);
  unsigned* krem   = (unsigned*)carve((size_t)BH * 4);
  float* thr       = (float*)carve((size_t)BH * 4);
  float* scores    = (float*)carve((size_t)BH * SEQ * SEQ * 4);   // 192 MB

  const int nx = BATCH * SEQ * DMODEL;
  k_cast  <<<(nx + 255) / 256, 256, 0, stream>>>(x, xb, nx);
  k_cast_T<<<(DMODEL * 3 * DMODEL + 255) / 256, 256, 0, stream>>>(Wqkv, wqkvT, DMODEL, 3 * DMODEL);
  k_cast_T<<<(DMODEL * DMODEL + 255) / 256, 256, 0, stream>>>(Wout, woutT, DMODEL, DMODEL);

  k_gemm_qkv   <<<(256 * 36) / 4, 128, 0, stream>>>(xb, wqkvT, bqkv, qb, kb, vtb);
  k_gemm_scores<<<(BH * 1024) / 4, 128, 0, stream>>>(qb, kb, scores);

  k_sel_init<<<BH, 256, 0, stream>>>(hist, prefix, krem);
  for (int pass = 0; pass < 4; ++pass) {
    k_sel_hist<<<BH * SPLIT, 256, 0, stream>>>(scores, prefix, hist, pass);
    k_sel_pick<<<BH, 256, 0, stream>>>(hist, prefix, krem, thr, pass);
  }

  k_softmax     <<<BH * SEQ, 256, 0, stream>>>(scores, thr);
  k_gemm_attnout<<<(BH * 64) / 4, 128, 0, stream>>>(scores, vtb, ob);
  k_gemm_final  <<<(256 * 12) / 4, 128, 0, stream>>>(ob, woutT, bout, out);
}
